// Attention_24326694764565
// MI455X (gfx1250) — matrix operand order repaired and checked by reading, not compiled, not run on hardware
//
#include <hip/hip_runtime.h>

#define BATCH   8
#define SEQ     1024
#define CDIM    1024
#define NHEADS  16
#define HDIM    64
#define ATT_SCALE 0.125f      // 64^-0.5
#define NEGV    -1e9f

typedef __attribute__((ext_vector_type(16))) __bf16 v16bf;
typedef __attribute__((ext_vector_type(8)))  float  v8f;

union Frag16 {
    v16bf        v;
    unsigned int u32[8];
    uint4        u128[2];
};

// Native bf16 convert (RNE) -> v_cvt_pk_bf16_f32 on gfx1250
__device__ __forceinline__ unsigned short f2bf(float f) {
    __bf16 h = (__bf16)f;
    return __builtin_bit_cast(unsigned short, h);
}

// K index inside a 16x32 bf16 A-fragment: lanes 0-15 hold K {0..15},
// lanes 16-31 hold K {16..31}; 2 values per VGPR.
__device__ __forceinline__ int a_kidx(int vv, int half) {
    return half * 16 + vv * 2;
}

__device__ __forceinline__ v8f wmma_bf16(const Frag16& a, const Frag16& b, v8f c) {
    return __builtin_amdgcn_wmma_f32_16x16x32_bf16(false, a.v, false, b.v,
                                                   (short)0, c, false, false);
}

#define LDA 40   // padded LDS strides (bf16 elements); 80B rows keep 16B alignment
#define LDB 40

// ---------------------------------------------------------------------------
// Y(bf16, scattered to (B,H,N,D)) = X(fp32, R x 1024) @ W(fp32, 1024 x 1024)
// Block: 128 threads (4 waves). Tile 128x64; each wave owns 32 rows x 64 cols
// (two 16-row strips sharing the B fragments) -> 8 WMMA per 32-deep k-step.
// ---------------------------------------------------------------------------
__global__ __launch_bounds__(128)
void proj_gemm_kernel(const float* __restrict__ X, const float* __restrict__ W,
                      unsigned short* __restrict__ Yh) {
    __shared__ __align__(16) unsigned short Asl[128 * LDA];
    __shared__ __align__(16) unsigned short Btl[64 * LDB];   // transposed: [col][k]
    const int tid  = threadIdx.x;
    const int lane = tid & 31;
    const int wave = tid >> 5;
    const int half = lane >> 4;
    const int nn   = lane & 15;
    const int rowBase = blockIdx.x * 128;
    const int colBase = blockIdx.y * 64;

    v8f acc[2][4] = {};

    for (int k0 = 0; k0 < CDIM; k0 += 32) {
        __syncthreads();
#pragma unroll
        for (int i = 0; i < 32; ++i) {                 // 128x32 A tile, fp32 -> bf16
            int idx = tid + i * 128;
            int r = idx >> 5, kk = idx & 31;
            float x = __builtin_nontemporal_load(&X[(size_t)(rowBase + r) * CDIM + k0 + kk]);
            Asl[r * LDA + kk] = f2bf(x);
        }
#pragma unroll
        for (int i = 0; i < 16; ++i) {                 // 32x64 B tile -> transposed LDS
            int idx = tid + i * 128;
            int kk = idx >> 6, c = idx & 63;           // coalesced over c
            Btl[c * LDB + kk] = f2bf(W[(size_t)(k0 + kk) * CDIM + colBase + c]);
        }
        __syncthreads();

        Frag16 a[2], b[4];                             // preload everything, then WMMA burst
#pragma unroll
        for (int s = 0; s < 2; ++s) {
            const unsigned short* ar = &Asl[(wave * 32 + s * 16 + nn) * LDA];
#pragma unroll
            for (int vv = 0; vv < 8; ++vv)
                a[s].u32[vv] = *(const unsigned int*)(ar + a_kidx(vv, half));
        }
#pragma unroll
        for (int t = 0; t < 4; ++t) {
            const uint4* p = (const uint4*)&Btl[(t * 16 + nn) * LDB + half * 16];
            b[t].u128[0] = p[0]; b[t].u128[1] = p[1];
        }
#pragma unroll
        for (int s = 0; s < 2; ++s)
#pragma unroll
            for (int t = 0; t < 4; ++t)
                acc[s][t] = wmma_bf16(a[s], b[t], acc[s][t]);
    }

    // C layout: VGPR r -> row r + 8*half, lane%16 -> col. Scatter to (B,H,N,D).
#pragma unroll
    for (int s = 0; s < 2; ++s)
#pragma unroll
    for (int t = 0; t < 4; ++t) {
        int col = colBase + t * 16 + nn;
        int h = col >> 6, d = col & 63;
#pragma unroll
        for (int r = 0; r < 8; ++r) {
            int row = rowBase + wave * 32 + s * 16 + r + 8 * half;
            int bb = row >> 10, n = row & 1023;
            Yh[((size_t)(bb * NHEADS + h) * SEQ + n) * HDIM + d] = f2bf(acc[s][t][r]);
        }
    }
}

// ---------------------------------------------------------------------------
// out(fp32) = Xh(bf16, R x 1024) @ W(fp32) + bias   (same 128x64 tiling)
// ---------------------------------------------------------------------------
__global__ __launch_bounds__(128)
void out_gemm_kernel(const unsigned short* __restrict__ Xh, const float* __restrict__ W,
                     const float* __restrict__ bias, float* __restrict__ out) {
    __shared__ __align__(16) unsigned short Asl[128 * LDA];
    __shared__ __align__(16) unsigned short Btl[64 * LDB];
    const int tid  = threadIdx.x;
    const int lane = tid & 31;
    const int wave = tid >> 5;
    const int half = lane >> 4;
    const int nn   = lane & 15;
    const int rowBase = blockIdx.x * 128;
    const int colBase = blockIdx.y * 64;

    v8f acc[2][4] = {};

    for (int k0 = 0; k0 < CDIM; k0 += 32) {
        __syncthreads();
#pragma unroll
        for (int i = 0; i < 32; ++i) {
            int idx = tid + i * 128;
            int r = idx >> 5, kk = idx & 31;
            Asl[r * LDA + kk] =
                __builtin_nontemporal_load(&Xh[(size_t)(rowBase + r) * CDIM + k0 + kk]);
        }
#pragma unroll
        for (int i = 0; i < 16; ++i) {
            int idx = tid + i * 128;
            int kk = idx >> 6, c = idx & 63;
            Btl[c * LDB + kk] = f2bf(W[(size_t)(k0 + kk) * CDIM + colBase + c]);
        }
        __syncthreads();

        Frag16 a[2], b[4];
#pragma unroll
        for (int s = 0; s < 2; ++s) {
            const unsigned short* ar = &Asl[(wave * 32 + s * 16 + nn) * LDA];
#pragma unroll
            for (int vv = 0; vv < 8; ++vv)
                a[s].u32[vv] = *(const unsigned int*)(ar + a_kidx(vv, half));
        }
#pragma unroll
        for (int t = 0; t < 4; ++t) {
            const uint4* p = (const uint4*)&Btl[(t * 16 + nn) * LDB + half * 16];
            b[t].u128[0] = p[0]; b[t].u128[1] = p[1];
        }
#pragma unroll
        for (int s = 0; s < 2; ++s)
#pragma unroll
            for (int t = 0; t < 4; ++t)
                acc[s][t] = wmma_bf16(a[s], b[t], acc[s][t]);
    }

#pragma unroll
    for (int s = 0; s < 2; ++s)
#pragma unroll
    for (int t = 0; t < 4; ++t) {
        int col = colBase + t * 16 + nn;
        float bcol = bias[col];
#pragma unroll
        for (int r = 0; r < 8; ++r) {
            int row = rowBase + wave * 32 + s * 16 + r + 8 * half;
            __builtin_nontemporal_store(acc[s][t][r] + bcol, &out[(size_t)row * CDIM + col]);
        }
    }
}

// ---------------------------------------------------------------------------
// Flash attention: grid (N/64, H, B), 128 threads (4 waves x 16 queries).
// Qh/Kh/Vh bf16 in (B,H,SEQ,HDIM). Output Xh bf16 in (B,N,C).
// ---------------------------------------------------------------------------
__global__ __launch_bounds__(128)
void flash_attn_kernel(const unsigned short* __restrict__ Qh,
                       const unsigned short* __restrict__ Kh,
                       const unsigned short* __restrict__ Vh,
                       const int* __restrict__ kmask,    // (B,N): key mask
                       const int* __restrict__ tmask,    // (B,N,M)
                       unsigned short* __restrict__ Xh) {
    __shared__ __align__(16) unsigned short Kl[32 * 72];      // [key][dim]
    __shared__ __align__(16) unsigned short Vt[64 * 40];      // [dim][key]
    __shared__ __align__(16) unsigned short Pl[4 * 16 * 40];  // per-wave P staging

    const int tid  = threadIdx.x;
    const int lane = tid & 31;
    const int wave = tid >> 5;
    const int half = lane >> 4;
    const int nn   = lane & 15;
    const int b = blockIdx.z;
    const int h = blockIdx.y;
    const int qbase = blockIdx.x * 64 + wave * 16;
    const size_t bh = (size_t)(b * NHEADS + h) * SEQ;

    // Q A-fragments for D = [0,32) and [32,64) live in VGPRs for the whole pass
    Frag16 a0, a1;
    {
        const unsigned short* qrow = Qh + (bh + qbase + nn) * HDIM;
#pragma unroll
        for (int vv = 0; vv < 8; ++vv) {
            int kk = a_kidx(vv, half);
            a0.u32[vv] = *(const unsigned int*)(qrow + kk);
            a1.u32[vv] = *(const unsigned int*)(qrow + 32 + kk);
        }
    }

    float mrun[8], lrun[8];
    v8f o[4] = {};
#pragma unroll
    for (int r = 0; r < 8; ++r) { mrun[r] = -INFINITY; lrun[r] = 0.f; }

    for (int j = 0; j < SEQ; j += 32) {
        __syncthreads();
#pragma unroll
        for (int i = 0; i < 16; ++i) {       // stage 32 keys x 64 dims of K and V
            int idx = tid + i * 128;
            int key = idx >> 6, d = idx & 63;
            Kl[key * 72 + d] = Kh[(bh + j + key) * HDIM + d];
            Vt[d * 40 + key] = Vh[(bh + j + key) * HDIM + d];
        }
        if (j + 32 < SEQ) {                  // pull next tile toward L2/L0
            int key = tid >> 6, d = tid & 63;
            __builtin_prefetch(&Kh[(bh + j + 32 + key) * HDIM + d], 0, 0);
            __builtin_prefetch(&Vh[(bh + j + 32 + key) * HDIM + d], 0, 0);
        }
        __syncthreads();

        // S = Q K^T: preload all 4 K fragments, then a WMMA burst
        Frag16 bk[4];
#pragma unroll
        for (int c = 0; c < 2; ++c) {
            const uint4* p0 = (const uint4*)&Kl[nn * 72 + c * 32 + half * 16];
            bk[c].u128[0] = p0[0]; bk[c].u128[1] = p0[1];
            const uint4* p1 = (const uint4*)&Kl[(16 + nn) * 72 + c * 32 + half * 16];
            bk[2 + c].u128[0] = p1[0]; bk[2 + c].u128[1] = p1[1];
        }
        v8f c0 = {}, c1 = {};
        c0 = wmma_bf16(a0, bk[0], c0);
        c0 = wmma_bf16(a1, bk[1], c0);
        c1 = wmma_bf16(a0, bk[2], c1);
        c1 = wmma_bf16(a1, bk[3], c1);

        // masks + scale + online softmax (rows in VGPR index r, cols in lanes)
        int km0 = kmask[b * SEQ + j + nn];
        int km1 = kmask[b * SEQ + j + 16 + nn];
        float p0[8], p1[8];
#pragma unroll
        for (int r = 0; r < 8; ++r) {
            int q = qbase + r + 8 * half;
            const int* tr = tmask + ((size_t)b * SEQ + q) * SEQ + j;
            int tm0 = __builtin_nontemporal_load(&tr[nn]);
            int tm1 = __builtin_nontemporal_load(&tr[16 + nn]);
            float s0 = (km0 == 0 || tm0 == 0) ? NEGV : c0[r] * ATT_SCALE;
            float s1 = (km1 == 0 || tm1 == 0) ? NEGV : c1[r] * ATT_SCALE;
            float mx = fmaxf(s0, s1);
            mx = fmaxf(mx, __shfl_xor(mx, 1, 32));
            mx = fmaxf(mx, __shfl_xor(mx, 2, 32));
            mx = fmaxf(mx, __shfl_xor(mx, 4, 32));
            mx = fmaxf(mx, __shfl_xor(mx, 8, 32));
            float mnew = fmaxf(mrun[r], mx);
            float e0 = __expf(s0 - mnew);
            float e1 = __expf(s1 - mnew);
            float ls = e0 + e1;
            ls += __shfl_xor(ls, 1, 32);
            ls += __shfl_xor(ls, 2, 32);
            ls += __shfl_xor(ls, 4, 32);
            ls += __shfl_xor(ls, 8, 32);
            float alpha = __expf(mrun[r] - mnew);
            mrun[r] = mnew;
            lrun[r] = lrun[r] * alpha + ls;
#pragma unroll
            for (int t = 0; t < 4; ++t) o[t][r] *= alpha;
            p0[r] = e0; p1[r] = e1;
        }

        // C-layout -> A-layout shuffle of P through wave-private LDS
        unsigned short* pw = Pl + wave * (16 * 40);
#pragma unroll
        for (int r = 0; r < 8; ++r) {
            int m = r + 8 * half;
            pw[m * 40 + nn]      = f2bf(p0[r]);
            pw[m * 40 + 16 + nn] = f2bf(p1[r]);
        }
        asm volatile("s_wait_dscnt 0" ::: "memory");   // wave-local LDS visibility

        Frag16 pa, bv[4];
        {
            const unsigned short* pr = pw + nn * 40;
#pragma unroll
            for (int vv = 0; vv < 8; ++vv)
                pa.u32[vv] = *(const unsigned int*)(pr + a_kidx(vv, half));
        }
#pragma unroll
        for (int t = 0; t < 4; ++t) {
            const uint4* p = (const uint4*)&Vt[(t * 16 + nn) * 40 + half * 16];
            bv[t].u128[0] = p[0]; bv[t].u128[1] = p[1];
        }
#pragma unroll
        for (int t = 0; t < 4; ++t)          // O += P @ V, one 16-dim chunk per t
            o[t] = wmma_bf16(pa, bv[t], o[t]);
    }

#pragma unroll
    for (int r = 0; r < 8; ++r) lrun[r] = 1.0f / lrun[r];
#pragma unroll
    for (int t = 0; t < 4; ++t) {
        int d = h * HDIM + t * 16 + nn;
#pragma unroll
        for (int r = 0; r < 8; ++r) {
            int q = qbase + r + 8 * half;
            Xh[((size_t)b * SEQ + q) * CDIM + d] = f2bf(o[t][r] * lrun[r]);
        }
    }
}

// ---------------------------------------------------------------------------
// Harness entry. Workspace layout (bf16/ushort): Qh | Kh | Vh | Xh = 64 MB.
// ---------------------------------------------------------------------------
extern "C" void kernel_launch(void* const* d_in, const int* in_sizes, int n_in,
                              void* d_out, int out_size, void* d_ws, size_t ws_size,
                              hipStream_t stream) {
    const float* q     = (const float*)d_in[0];
    const float* k     = (const float*)d_in[1];
    const float* v     = (const float*)d_in[2];
    const int*   mask  = (const int*)d_in[3];
    const int*   tmask = (const int*)d_in[4];
    const float* Wq    = (const float*)d_in[5];
    const float* Wk    = (const float*)d_in[6];
    const float* Wv    = (const float*)d_in[7];
    const float* Wo    = (const float*)d_in[8];
    const float* bo    = (const float*)d_in[9];
    float* out = (float*)d_out;

    unsigned short* ws = (unsigned short*)d_ws;
    const size_t nE = (size_t)BATCH * NHEADS * SEQ * HDIM;   // 8M elements
    unsigned short* Qh = ws;
    unsigned short* Kh = ws + nE;
    unsigned short* Vh = ws + 2 * nE;
    unsigned short* Xh = ws + 3 * nE;

    dim3 gblk(128);
    dim3 ggemm(BATCH * SEQ / 128, CDIM / 64);                // (64, 16)

    proj_gemm_kernel<<<ggemm, gblk, 0, stream>>>(q, Wq, Qh);
    proj_gemm_kernel<<<ggemm, gblk, 0, stream>>>(k, Wk, Kh);
    proj_gemm_kernel<<<ggemm, gblk, 0, stream>>>(v, Wv, Vh);

    flash_attn_kernel<<<dim3(SEQ / 64, NHEADS, BATCH), gblk, 0, stream>>>(
        Qh, Kh, Vh, mask, tmask, Xh);

    out_gemm_kernel<<<ggemm, gblk, 0, stream>>>(Xh, Wo, bo, out);
}
